// NodeNetwork_70265664963043
// MI455X (gfx1250) — matrix operand order repaired and checked by reading, not compiled, not run on hardware
//
#include <hip/hip_runtime.h>
#include <hip/hip_bf16.h>
#include <math.h>

typedef __attribute__((ext_vector_type(16))) _Float16 v16h;
typedef __attribute__((ext_vector_type(8)))  _Float16 v8h;
typedef __attribute__((ext_vector_type(8)))  float    v8f;

#define NB 8
#define NN 256
#define NF 32
#define OUTH 32
#define MIDE 48

// gfx1250 has a hardware V_TANH_F32 (TRANS pipe, co-executes with WMMA).
__device__ __forceinline__ float fast_tanh(float x) {
#if __has_builtin(__builtin_amdgcn_tanhf)
    return __builtin_amdgcn_tanhf(x);
#else
    return tanhf(x);
#endif
}

// ---------------------------------------------------------------------------
// Kernel 1: per-node precompute.
//   A_part[g,0:48] = node[g] @ eW1[0:32,:]
//   D_part[g,0:48] = node[g] @ eW1[32:64,:]
//   out1[g,0:32]   = tanh(relu(node[g]@n1W1+n1b1)@n1W2+n1b2)
// g = b*256+n, 2048 blocks x 64 threads.
// ---------------------------------------------------------------------------
__global__ void precompute_kernel(const float* __restrict__ node,
                                  const float* __restrict__ eW1,
                                  const float* __restrict__ n1W1,
                                  const float* __restrict__ n1b1,
                                  const float* __restrict__ n1W2,
                                  const float* __restrict__ n1b2,
                                  float* __restrict__ A_part,
                                  float* __restrict__ D_part,
                                  float* __restrict__ out1) {
    int g = blockIdx.x;
    int t = threadIdx.x;
    __shared__ float x[NF];
    __shared__ float mid[OUTH];
    if (t < NF) x[t] = node[g * NF + t];
    __syncthreads();
    if (t < MIDE) {
        float a = 0.f, d = 0.f;
        #pragma unroll 8
        for (int k = 0; k < NF; k++) {
            a += x[k] * eW1[k * MIDE + t];
            d += x[k] * eW1[(NF + k) * MIDE + t];
        }
        A_part[g * MIDE + t] = a;
        D_part[g * MIDE + t] = d;
    }
    if (t < OUTH) {
        float m = n1b1[t];
        #pragma unroll 8
        for (int k = 0; k < NF; k++) m += x[k] * n1W1[k * OUTH + t];
        mid[t] = fmaxf(m, 0.f);
    }
    __syncthreads();
    if (t < OUTH) {
        float o = n1b2[t];
        #pragma unroll 8
        for (int k = 0; k < OUTH; k++) o += mid[k] * n1W2[k * OUTH + t];
        out1[g * OUTH + t] = fast_tanh(o);
    }
}

// ---------------------------------------------------------------------------
// Kernel 2: one block per (b,i) group of 255 neighbors. 256 threads = 8 waves.
//   hidden[r] = relu(A_part[i] + D_part[j(r)] + edge*eW1[64] + eb1)  (f16, LDS)
//   h = tanh(hidden @ eW2 + eb2)  via v_wmma_f32_16x16x32_f16
//   msum = sum_r h[r]; out2 = node-MLP2(msum); write normalized concat.
// ---------------------------------------------------------------------------
__global__ void edge_kernel(const float* __restrict__ inp_edge,
                            const float* __restrict__ eW1,
                            const float* __restrict__ eb1,
                            const float* __restrict__ eW2,
                            const float* __restrict__ eb2,
                            const float* __restrict__ n2W1,
                            const float* __restrict__ n2b1,
                            const float* __restrict__ n2W2,
                            const float* __restrict__ n2b2,
                            const float* __restrict__ A_part,
                            const float* __restrict__ D_part,
                            const float* __restrict__ out1,
                            float* __restrict__ out) {
    const int g   = blockIdx.x;          // b*256 + i
    const int bat = g >> 8;
    const int i   = g & 255;
    const int tid = threadIdx.x;

    __shared__ __align__(32) _Float16 hid[256][64];   // 255 rows + 1 pad, K padded 48->64
    __shared__ __align__(32) _Float16 w2T[32][64];    // eW2 column-major, K padded
    __shared__ float Ai[MIDE], w64[MIDE], eb1s[MIDE], eb2s[OUTH];
    __shared__ float msum[OUTH], mid2[OUTH], obuf[2 * OUTH];
    __shared__ float snorm;

    if (tid < MIDE) {
        Ai[tid]   = A_part[g * MIDE + tid];
        w64[tid]  = eW1[64 * MIDE + tid];
        eb1s[tid] = eb1[tid];
    }
    if (tid < OUTH) { eb2s[tid] = eb2[tid]; msum[tid] = 0.f; }
    if (tid == 0) snorm = 0.f;
    // stage eW2 transposed: w2T[col][k]
    for (int idx = tid; idx < 32 * 64; idx += 256) {
        int col = idx >> 6, k = idx & 63;
        w2T[col][k] = (k < MIDE) ? (_Float16)eW2[k * OUTH + col] : (_Float16)0.f;
    }
    __syncthreads();

    // stage hidden activations (one row per thread)
    {
        int r = tid;
        if (r < 255) {
            int j = (r < i) ? r : r + 1;
            float ev = inp_edge[g * 255 + r];
            const float* Dp = D_part + (bat * NN + j) * MIDE;
            #pragma unroll 8
            for (int t = 0; t < MIDE; t++) {
                float h = Ai[t] + Dp[t] + ev * w64[t] + eb1s[t];
                hid[r][t] = (_Float16)fmaxf(h, 0.f);
            }
            #pragma unroll
            for (int t = MIDE; t < 64; t++) hid[r][t] = (_Float16)0.f;
        } else {
            #pragma unroll
            for (int t = 0; t < 64; t++) hid[255][t] = (_Float16)0.f;
        }
    }
    __syncthreads();

    const int lane = tid & 31;
    const int wave = tid >> 5;
    const int hf   = lane >> 4;     // lane half
    const int lcol = lane & 15;

    // Each wave owns 2 M-tiles; 2 N-tiles x 2 K-steps each -> 8 WMMAs/wave.
    for (int m = 0; m < 2; m++) {
        const int mt = wave * 2 + m;
        v8f acc0 = {};   // cols 0..15
        v8f acc1 = {};   // cols 16..31
        #pragma unroll
        for (int ks = 0; ks < 2; ks++) {
            // A fragment: row = mt*16 + lane%16, K = ks*32 + hf*16 + e  (contiguous)
            v16h a = *(const v16h*)&hid[mt * 16 + lcol][ks * 32 + hf * 16];
            // B fragment: col = nt*16 + lane%16, K = ks*32 + hf*16 + e  (contiguous)
            v16h bf0 = *(const v16h*)&w2T[lcol][ks * 32 + hf * 16];
            v16h bf1 = *(const v16h*)&w2T[16 + lcol][ks * 32 + hf * 16];
            acc0 = __builtin_amdgcn_wmma_f32_16x16x32_f16(
                       false, a, false, bf0, (short)0, acc0, false, false);
            acc1 = __builtin_amdgcn_wmma_f32_16x16x32_f16(
                       false, a, false, bf1, (short)0, acc1, false, false);
        }
        // bias + tanh + partial neighbor-sum (C layout: row = mt*16 + hf*8 + v).
        // Branch-free mask for the single padding row (grow == 255).
        float p0 = 0.f, p1 = 0.f;
        #pragma unroll
        for (int v = 0; v < 8; v++) {
            int grow = mt * 16 + hf * 8 + v;
            float msk = (grow < 255) ? 1.f : 0.f;
            p0 += msk * fast_tanh(acc0[v] + eb2s[lcol]);
            p1 += msk * fast_tanh(acc1[v] + eb2s[16 + lcol]);
        }
        atomicAdd(&msum[lcol], p0);
        atomicAdd(&msum[16 + lcol], p1);
    }
    __syncthreads();

    // node MLP2 on message sum
    if (tid < OUTH) {
        float m2 = n2b1[tid];
        #pragma unroll 8
        for (int k = 0; k < OUTH; k++) m2 += msum[k] * n2W1[k * OUTH + tid];
        mid2[tid] = fmaxf(m2, 0.f);
    }
    __syncthreads();
    if (tid < OUTH) {
        float o2 = n2b2[tid];
        #pragma unroll 8
        for (int k = 0; k < OUTH; k++) o2 += mid2[k] * n2W2[k * OUTH + tid];
        obuf[OUTH + tid] = fast_tanh(o2);
        obuf[tid]        = out1[g * OUTH + tid];
    }
    __syncthreads();
    if (tid < 2 * OUTH) atomicAdd(&snorm, obuf[tid] * obuf[tid]);
    __syncthreads();
    if (tid < 2 * OUTH) out[g * 2 * OUTH + tid] = obuf[tid] * rsqrtf(snorm);
}

extern "C" void kernel_launch(void* const* d_in, const int* in_sizes, int n_in,
                              void* d_out, int out_size, void* d_ws, size_t ws_size,
                              hipStream_t stream) {
    const float* inp_node = (const float*)d_in[0];
    const float* inp_edge = (const float*)d_in[1];
    const float* eW1  = (const float*)d_in[2];
    const float* eb1  = (const float*)d_in[3];
    const float* eW2  = (const float*)d_in[4];
    const float* eb2  = (const float*)d_in[5];
    const float* n1W1 = (const float*)d_in[6];
    const float* n1b1 = (const float*)d_in[7];
    const float* n1W2 = (const float*)d_in[8];
    const float* n1b2 = (const float*)d_in[9];
    const float* n2W1 = (const float*)d_in[10];
    const float* n2b1 = (const float*)d_in[11];
    const float* n2W2 = (const float*)d_in[12];
    const float* n2b2 = (const float*)d_in[13];
    float* out = (float*)d_out;

    float* ws     = (float*)d_ws;
    float* A_part = ws;                                 // 2048*48 floats
    float* D_part = A_part + (NB * NN) * MIDE;          // 2048*48 floats
    float* out1   = D_part + (NB * NN) * MIDE;          // 2048*32 floats

    precompute_kernel<<<NB * NN, 64, 0, stream>>>(
        inp_node, eW1, n1W1, n1b1, n1W2, n1b2, A_part, D_part, out1);

    edge_kernel<<<NB * NN, 256, 0, stream>>>(
        inp_edge, eW1, eb1, eW2, eb2, n2W1, n2b1, n2W2, n2b2,
        A_part, D_part, out1, out);
}
